// Addresser_50130858279095
// MI455X (gfx1250) — matrix operand order repaired and checked by reading, not compiled, not run on hardware
//
#include <hip/hip_runtime.h>

typedef __attribute__((ext_vector_type(16))) __bf16 v16bf;
typedef __attribute__((ext_vector_type(8)))  float  v8f;
typedef __attribute__((ext_vector_type(2)))  float  v2f;

#define D_MODEL 2048
#define N_ADDR  512
#define N_TOT   1024
#define TM      16
#define XPITCH  2064   // fused fallback: bf16 elems per X row in LDS
#define WPITCH  40     // fused fallback: bf16 elems per W column in LDS
#define RPITCH  520    // f32 elems per result row in LDS

#if defined(__has_builtin)
#  if __has_builtin(__builtin_amdgcn_wmma_f32_16x16x4_f32)
#    define HAVE_WMMA_F32X4 1
#  endif
#endif

struct P1 { unsigned short x[TM * XPITCH]; unsigned short w[N_TOT * WPITCH]; };
struct P2 { float addr[TM * RPITCH]; float data[TM * RPITCH]; };
union  SU { P1 p1; P2 p2; };

union VB { uint4 q[2]; v16bf v; };

__device__ __forceinline__ unsigned short f2bf(float f) {
  unsigned u = __builtin_bit_cast(unsigned, f);
  u += 0x7FFFu + ((u >> 16) & 1u);            // round-to-nearest-even
  return (unsigned short)(u >> 16);
}

// ---------------------------------------------------------------------------
// Pre-pass 1: W (2048 x 512 f32, addr||data) -> wt[n][k] bf16, n-major (4 MB)
// ---------------------------------------------------------------------------
__global__ __launch_bounds__(256)
void convert_w_kernel(const float* __restrict__ Wa, const float* __restrict__ Wd,
                      unsigned short* __restrict__ wt) {
  int idx = blockIdx.x * 256 + threadIdx.x;       // 0 .. N_TOT*D_MODEL-1
  int n = idx & (N_TOT - 1);                      // coalesced read over n
  int k = idx >> 10;
  float v = (n < N_ADDR) ? Wa[(size_t)k * N_ADDR + n]
                         : Wd[(size_t)k * N_ADDR + (n - N_ADDR)];
  wt[(size_t)n * D_MODEL + k] = f2bf(v);
}

// ---------------------------------------------------------------------------
// Pre-pass 2: X f32 -> xb bf16 (row-major, coalesced)
// ---------------------------------------------------------------------------
__global__ __launch_bounds__(256)
void convert_x_kernel(const float* __restrict__ X, unsigned short* __restrict__ xb) {
  int idx = blockIdx.x * 256 + threadIdx.x;       // one float4 each
  float4 v = *(const float4*)&X[(size_t)idx * 4];
  unsigned lo = (unsigned)f2bf(v.x) | ((unsigned)f2bf(v.y) << 16);
  unsigned hi = (unsigned)f2bf(v.z) | ((unsigned)f2bf(v.w) << 16);
  *(uint2*)&xb[(size_t)idx * 4] = make_uint2(lo, hi);
}

// ---------------------------------------------------------------------------
// Main kernel: barrier-free bf16 WMMA GEMM from global, register softmax,
// LDS exchange, per-token outer product via f32 16x16x4 WMMA.
// ---------------------------------------------------------------------------
__global__ __launch_bounds__(256)
void addresser_main(const unsigned short* __restrict__ xb,
                    const unsigned short* __restrict__ wt,
                    const float* __restrict__ ba, const float* __restrict__ bd,
                    float* __restrict__ out) {
  __shared__ P2 s;
  const int tid  = threadIdx.x;
  const int lane = tid & 31;
  const int wv   = tid >> 5;           // waves 0-3 -> addr cols, 4-7 -> data cols
  const int m0   = blockIdx.x * TM;

  const int rl    = lane & 15;
  const int khalf = (lane < 16) ? 0 : 8;    // A-frag K sub-offset (16-bit A 16x32 layout)
  const int ksel  = (lane < 16) ? 0 : 8;    // B-frag K sub-offset (table order, matches A)
  const int mhi   = (lane >> 4) * 8;        // C/D layout row offset

  const unsigned short* xrow = xb + (size_t)(m0 + rl) * D_MODEL;
  // Per-lane base pointers for the 8 owned output columns (B is [n][k] bf16).
  const unsigned short* wp0 = wt + (size_t)((wv * 8) * 16 + rl) * D_MODEL + ksel;

  v8f acc[8] = {};

  for (int ks = 0; ks < D_MODEL / 32; ++ks) {
    const int k0 = ks * 32;
    VB a;
    a.q[0] = *(const uint4*)(xrow + k0 + khalf);        // K k0+khalf..+7
    a.q[1] = *(const uint4*)(xrow + k0 + khalf + 16);   // K k0+16+khalf..+7
    if (ks + 1 < D_MODEL / 32) __builtin_prefetch(xrow + k0 + 32, 0, 0);

    #pragma unroll
    for (int j = 0; j < 8; ++j) {
      const unsigned short* wp = wp0 + (size_t)j * 16 * D_MODEL + k0;
      VB b;
      b.q[0] = *(const uint4*)wp;           // K k0+ksel..+7
      b.q[1] = *(const uint4*)(wp + 16);    // K k0+16+ksel..+7
      if (ks + 1 < D_MODEL / 32) __builtin_prefetch(wp + 32, 0, 0);
      acc[j] = __builtin_amdgcn_wmma_f32_16x16x32_bf16(
          false, a.v, false, b.v, (short)0, acc[j], false, false);
    }
  }

  // ---- bias ----
  #pragma unroll
  for (int j = 0; j < 8; ++j) {
    int col = ((wv & 3) * 8 + j) * 16 + rl;
    float bb = (wv < 4) ? ba[col] : bd[col];
    #pragma unroll
    for (int v = 0; v < 8; ++v) acc[j][v] += bb;
  }

  // ---- softmax over each head's 64 addr columns (waves 0-3 only) ----
  if (wv < 4) {
    #pragma unroll
    for (int hh = 0; hh < 2; ++hh) {
      #pragma unroll
      for (int v = 0; v < 8; ++v) {
        float mx = fmaxf(fmaxf(acc[hh*4+0][v], acc[hh*4+1][v]),
                         fmaxf(acc[hh*4+2][v], acc[hh*4+3][v]));
        mx = fmaxf(mx, __shfl_xor(mx, 1, 32));
        mx = fmaxf(mx, __shfl_xor(mx, 2, 32));
        mx = fmaxf(mx, __shfl_xor(mx, 4, 32));
        mx = fmaxf(mx, __shfl_xor(mx, 8, 32));   // within 16-lane row group
        float ssum = 0.f;
        #pragma unroll
        for (int t = 0; t < 4; ++t) {
          float e = __expf(acc[hh*4+t][v] - mx);
          acc[hh*4+t][v] = e;
          ssum += e;
        }
        ssum += __shfl_xor(ssum, 1, 32);
        ssum += __shfl_xor(ssum, 2, 32);
        ssum += __shfl_xor(ssum, 4, 32);
        ssum += __shfl_xor(ssum, 8, 32);
        float inv = 1.0f / ssum;
        #pragma unroll
        for (int t = 0; t < 4; ++t) acc[hh*4+t][v] *= inv;
      }
    }
  }

  // ---- spill to LDS for cross-wave exchange ----
  {
    float* dst = (wv < 4) ? s.addr : s.data;
    #pragma unroll
    for (int j = 0; j < 8; ++j) {
      int col = ((wv & 3) * 8 + j) * 16 + rl;
      #pragma unroll
      for (int v = 0; v < 8; ++v)
        dst[(v + mhi) * RPITCH + col] = acc[j][v];
    }
  }
  __syncthreads();

  // ---- per-token outer product: addr^T(64x8) @ data(8x64) ----
  const int kb = (lane < 16) ? 0 : 2;
  #pragma unroll
  for (int t = 0; t < 2; ++t) {
    int m = wv * 2 + t;
    const float* ar = &s.addr[m * RPITCH];
    const float* dr = &s.data[m * RPITCH];
    float* orow = out + (size_t)(m0 + m) * 4096;
    #pragma unroll
    for (int ta = 0; ta < 4; ++ta) {
      int a0 = ta * 16;
#ifdef HAVE_WMMA_F32X4
      v2f af0, af4;
      af0.x = ar[(kb + 0) * 64 + a0 + rl]; af0.y = ar[(kb + 1) * 64 + a0 + rl];
      af4.x = ar[(kb + 4) * 64 + a0 + rl]; af4.y = ar[(kb + 5) * 64 + a0 + rl];
#endif
      #pragma unroll
      for (int td = 0; td < 4; ++td) {
        int d0 = td * 16;
        v8f o = {};
#ifdef HAVE_WMMA_F32X4
        v2f bf0, bf4;
        bf0.x = dr[(kb + 0) * 64 + d0 + rl]; bf0.y = dr[(kb + 1) * 64 + d0 + rl];
        bf4.x = dr[(kb + 4) * 64 + d0 + rl]; bf4.y = dr[(kb + 5) * 64 + d0 + rl];
        o = __builtin_amdgcn_wmma_f32_16x16x4_f32(false, af0, false, bf0, (short)0, o, false, false);
        o = __builtin_amdgcn_wmma_f32_16x16x4_f32(false, af4, false, bf4, (short)0, o, false, false);
#else
        #pragma unroll
        for (int v = 0; v < 8; ++v) {
          float sum = 0.f;
          #pragma unroll
          for (int h = 0; h < 8; ++h)
            sum += ar[h * 64 + a0 + mhi + v] * dr[h * 64 + d0 + rl];
          o[v] = sum;
        }
#endif
        #pragma unroll
        for (int v = 0; v < 8; ++v)
          orow[(a0 + mhi + v) * 64 + d0 + rl] = o[v];
      }
    }
  }
}

// ---------------------------------------------------------------------------
// Fused fallback (Round-1 kernel): used only if workspace is too small.
// ---------------------------------------------------------------------------
__global__ __launch_bounds__(256)
void addresser_fused(const float* __restrict__ X,
                     const float* __restrict__ Wa, const float* __restrict__ ba,
                     const float* __restrict__ Wd, const float* __restrict__ bd,
                     float* __restrict__ out) {
  __shared__ SU s;
  const int tid  = threadIdx.x;
  const int lane = tid & 31;
  const int wv   = tid >> 5;
  const int m0   = blockIdx.x * TM;

  for (int idx = tid; idx < TM * (D_MODEL / 4); idx += 256) {
    int r = idx / (D_MODEL / 4);
    int c = (idx % (D_MODEL / 4)) * 4;
    float4 v = *(const float4*)&X[(size_t)(m0 + r) * D_MODEL + c];
    unsigned lo = (unsigned)f2bf(v.x) | ((unsigned)f2bf(v.y) << 16);
    unsigned hi = (unsigned)f2bf(v.z) | ((unsigned)f2bf(v.w) << 16);
    *(uint2*)&s.p1.x[r * XPITCH + c] = make_uint2(lo, hi);
  }

  v8f acc[8] = {};
  const int rl    = lane & 15;
  const int khalf = (lane < 16) ? 0 : 8;
  const int ksel  = (lane < 16) ? 0 : 8;

  for (int ks = 0; ks < D_MODEL / 32; ++ks) {
    const int k0 = ks * 32;
    for (int idx = tid; idx < 32 * (N_TOT / 4); idx += 256) {
      int kk = idx >> 8;
      int n  = (idx & 255) * 4;
      const float* src = (n < N_ADDR)
          ? &Wa[(size_t)(k0 + kk) * N_ADDR + n]
          : &Wd[(size_t)(k0 + kk) * N_ADDR + (n - N_ADDR)];
      float4 v = *(const float4*)src;
      s.p1.w[(n + 0) * WPITCH + kk] = f2bf(v.x);
      s.p1.w[(n + 1) * WPITCH + kk] = f2bf(v.y);
      s.p1.w[(n + 2) * WPITCH + kk] = f2bf(v.z);
      s.p1.w[(n + 3) * WPITCH + kk] = f2bf(v.w);
    }
    __syncthreads();

    VB a;
    const unsigned short* xr = &s.p1.x[rl * XPITCH + k0 + khalf];
    a.q[0] = *(const uint4*)xr;
    a.q[1] = *(const uint4*)(xr + 16);

    #pragma unroll
    for (int j = 0; j < 8; ++j) {
      int n = (wv * 8 + j) * 16 + rl;
      VB b;
      const unsigned short* wp = &s.p1.w[n * WPITCH + ksel];
      b.q[0] = *(const uint4*)wp;
      b.q[1] = *(const uint4*)(wp + 16);
      acc[j] = __builtin_amdgcn_wmma_f32_16x16x32_bf16(
          false, a.v, false, b.v, (short)0, acc[j], false, false);
    }
    __syncthreads();
  }

  #pragma unroll
  for (int j = 0; j < 8; ++j) {
    int col = ((wv & 3) * 8 + j) * 16 + rl;
    float bb = (wv < 4) ? ba[col] : bd[col];
    #pragma unroll
    for (int v = 0; v < 8; ++v) acc[j][v] += bb;
  }

  if (wv < 4) {
    #pragma unroll
    for (int hh = 0; hh < 2; ++hh) {
      #pragma unroll
      for (int v = 0; v < 8; ++v) {
        float mx = fmaxf(fmaxf(acc[hh*4+0][v], acc[hh*4+1][v]),
                         fmaxf(acc[hh*4+2][v], acc[hh*4+3][v]));
        mx = fmaxf(mx, __shfl_xor(mx, 1, 32));
        mx = fmaxf(mx, __shfl_xor(mx, 2, 32));
        mx = fmaxf(mx, __shfl_xor(mx, 4, 32));
        mx = fmaxf(mx, __shfl_xor(mx, 8, 32));
        float ssum = 0.f;
        #pragma unroll
        for (int t = 0; t < 4; ++t) {
          float e = __expf(acc[hh*4+t][v] - mx);
          acc[hh*4+t][v] = e;
          ssum += e;
        }
        ssum += __shfl_xor(ssum, 1, 32);
        ssum += __shfl_xor(ssum, 2, 32);
        ssum += __shfl_xor(ssum, 4, 32);
        ssum += __shfl_xor(ssum, 8, 32);
        float inv = 1.0f / ssum;
        #pragma unroll
        for (int t = 0; t < 4; ++t) acc[hh*4+t][v] *= inv;
      }
    }
  }

  const int mhi = (lane >> 4) * 8;
  {
    float* dst = (wv < 4) ? s.p2.addr : s.p2.data;
    #pragma unroll
    for (int j = 0; j < 8; ++j) {
      int col = ((wv & 3) * 8 + j) * 16 + rl;
      #pragma unroll
      for (int v = 0; v < 8; ++v)
        dst[(v + mhi) * RPITCH + col] = acc[j][v];
    }
  }
  __syncthreads();

  const int kb = (lane < 16) ? 0 : 2;
  #pragma unroll
  for (int t = 0; t < 2; ++t) {
    int m = wv * 2 + t;
    const float* ar = &s.p2.addr[m * RPITCH];
    const float* dr = &s.p2.data[m * RPITCH];
    float* orow = out + (size_t)(m0 + m) * 4096;
    #pragma unroll
    for (int ta = 0; ta < 4; ++ta) {
      int a0 = ta * 16;
#ifdef HAVE_WMMA_F32X4
      v2f af0, af4;
      af0.x = ar[(kb + 0) * 64 + a0 + rl]; af0.y = ar[(kb + 1) * 64 + a0 + rl];
      af4.x = ar[(kb + 4) * 64 + a0 + rl]; af4.y = ar[(kb + 5) * 64 + a0 + rl];
#endif
      #pragma unroll
      for (int td = 0; td < 4; ++td) {
        int d0 = td * 16;
        v8f o = {};
#ifdef HAVE_WMMA_F32X4
        v2f bf0, bf4;
        bf0.x = dr[(kb + 0) * 64 + d0 + rl]; bf0.y = dr[(kb + 1) * 64 + d0 + rl];
        bf4.x = dr[(kb + 4) * 64 + d0 + rl]; bf4.y = dr[(kb + 5) * 64 + d0 + rl];
        o = __builtin_amdgcn_wmma_f32_16x16x4_f32(false, af0, false, bf0, (short)0, o, false, false);
        o = __builtin_amdgcn_wmma_f32_16x16x4_f32(false, af4, false, bf4, (short)0, o, false, false);
#else
        #pragma unroll
        for (int v = 0; v < 8; ++v) {
          float sum = 0.f;
          #pragma unroll
          for (int h = 0; h < 8; ++h)
            sum += ar[h * 64 + a0 + mhi + v] * dr[h * 64 + d0 + rl];
          o[v] = sum;
        }
#endif
        #pragma unroll
        for (int v = 0; v < 8; ++v)
          orow[(a0 + mhi + v) * 64 + d0 + rl] = o[v];
      }
    }
  }
}

extern "C" void kernel_launch(void* const* d_in, const int* in_sizes, int n_in,
                              void* d_out, int out_size, void* d_ws, size_t ws_size,
                              hipStream_t stream) {
  (void)n_in; (void)out_size;
  const float* X  = (const float*)d_in[0];
  const float* Wa = (const float*)d_in[1];
  const float* ba = (const float*)d_in[2];
  const float* Wd = (const float*)d_in[3];
  const float* bd = (const float*)d_in[4];
  float* out = (float*)d_out;
  const int tokens = in_sizes[0] / D_MODEL;     // 4096
  const int blocks = tokens / TM;               // 256

  const size_t wt_bytes = (size_t)N_TOT * D_MODEL * sizeof(unsigned short);   // 4 MB
  const size_t xb_bytes = (size_t)tokens * D_MODEL * sizeof(unsigned short);  // 16.8 MB

  if (ws_size >= wt_bytes + xb_bytes) {
    unsigned short* wt = (unsigned short*)d_ws;
    unsigned short* xb = (unsigned short*)((char*)d_ws + wt_bytes);
    hipLaunchKernelGGL(convert_w_kernel, dim3((N_TOT * D_MODEL) / 256), dim3(256), 0, stream,
                       Wa, Wd, wt);
    hipLaunchKernelGGL(convert_x_kernel, dim3((tokens * D_MODEL / 4) / 256), dim3(256), 0, stream,
                       X, xb);
    hipLaunchKernelGGL(addresser_main, dim3(blocks), dim3(256), 0, stream,
                       xb, wt, ba, bd, out);
  } else {
    hipLaunchKernelGGL(addresser_fused, dim3(blocks), dim3(256), 0, stream,
                       X, Wa, ba, Wd, bd, out);
  }
}
